// TFNLayer_85753317032378
// MI455X (gfx1250) — matrix operand order repaired and checked by reading, not compiled, not run on hardware
//
#include <hip/hip_runtime.h>
#include <hip/hip_bf16.h>
#include <math.h>

// ---------------------------------------------------------------------------
// TFN layer for MI455X (gfx1250, wave32, WMMA).
//
// Roofline: ~1 GB of traffic, 13 GFLOP of GEMM. out (25.6MB) + node_feat
// (25.6MB) are L2-resident (192MB L2) -> gather + atomic-scatter bound.
// Edge MLP done with v_wmma_f32_16x16x32_f16 (f32 accumulate); scatter via
// global_atomic_add_f32; self-connection kernel writes d_out directly
// (doubles as the init), then edge kernel accumulates atomically on top.
// ---------------------------------------------------------------------------

typedef __attribute__((ext_vector_type(16))) _Float16 v16h;
typedef __attribute__((ext_vector_type(8)))  _Float16 v8h;
typedef __attribute__((ext_vector_type(4)))  _Float16 v4h;
typedef __attribute__((ext_vector_type(8)))  float    v8f;
typedef __attribute__((ext_vector_type(4)))  float    v4f;

#define S2F        0.70710678118654752f   // 1/sqrt(2)
#define S2_3F      0.40824829046386302f   // 1/sqrt(6)
#define INV_SQRTM  0.17677669529663689f   // 1/sqrt(32)

__device__ __forceinline__ v8f wmma16(v16h a, v16h b, v8f c) {
    // D = A(16x32 f16) * B(32x16 f16) + C(16x16 f32)
    return __builtin_amdgcn_wmma_f32_16x16x32_f16(
        /*neg_a=*/false, a, /*neg_b=*/false, b,
        /*c_mod=*/(short)0, c, /*reuse_a=*/false, /*reuse_b=*/false);
}

// B fragment from LDS-resident transposed weights wT[N][Kdim].
// Lane L: N = nb*16 + L%16; K runs {k0..k0+7} and {k0+16..k0+23}, k0 = (L<16?0:8)+kOff
// -> two ds_load_b128.
__device__ __forceinline__ v16h lds_bfrag(const _Float16* wT, int Kdim,
                                          int nb, int kOff, int lane) {
    int n  = nb * 16 + (lane & 15);
    int k0 = ((lane & 16) ? 8 : 0) + kOff;
    const _Float16* p = wT + n * Kdim + k0;
    v8h lo = *(const v8h*)(p);
    v8h hi = *(const v8h*)(p + 16);
    return __builtin_shufflevector(lo, hi, 0,1,2,3,4,5,6,7,8,9,10,11,12,13,14,15);
}

// A fragment from LDS f16 matrix h[M][stride].
// Lane L = row M = L%16; K runs {k0..k0+7} and {k0+16..k0+23}, k0 = (L<16?0:8)+kOff.
__device__ __forceinline__ v16h lds_afrag(const _Float16* hb, int stride,
                                          int kOff, int lane) {
    int m  = lane & 15;
    int k0 = ((lane & 16) ? 8 : 0) + kOff;
    const _Float16* p = hb + m * stride + k0;
    v8h lo = *(const v8h*)(p);        // K = k0 .. k0+7
    v8h hi = *(const v8h*)(p + 16);   // K = k0+16 .. k0+23
    return __builtin_shufflevector(lo, hi, 0,1,2,3,4,5,6,7,8,9,10,11,12,13,14,15);
}

// A fragment (16x32) from a global f32 row, converting to f16 on the fly.
__device__ __forceinline__ v16h gl_afrag_f32(const float* row, int lane) {
    int k0 = (lane & 16) ? 8 : 0;
    v4f a0 = *(const v4f*)(row + k0);
    v4f a1 = *(const v4f*)(row + k0 + 4);
    v4f a2 = *(const v4f*)(row + k0 + 16);
    v4f a3 = *(const v4f*)(row + k0 + 20);
    v4h h0 = __builtin_convertvector(a0, v4h);
    v4h h1 = __builtin_convertvector(a1, v4h);
    v4h h2 = __builtin_convertvector(a2, v4h);
    v4h h3 = __builtin_convertvector(a3, v4h);
    v8h lo = __builtin_shufflevector(h0, h1, 0,1,2,3,4,5,6,7);
    v8h hi = __builtin_shufflevector(h2, h3, 0,1,2,3,4,5,6,7);
    return __builtin_shufflevector(lo, hi, 0,1,2,3,4,5,6,7,8,9,10,11,12,13,14,15);
}

__device__ __forceinline__ float silu(float x) {
    return x / (1.0f + __expf(-x));
}

// ---------------------------------------------------------------------------
// Kernel 1: self-connection, writes (does NOT accumulate) every element of
// out[n][128].  out[:, :32] = n0 @ sc_w0 * s ; out[:, 32+3v+j] = (n1_j @ sc_w1)*s.
// 160 threads = 5 waves; each wave owns one 16-node tile (50000/16/5 = 625 blocks).
// ---------------------------------------------------------------------------
__global__ __launch_bounds__(160) void tfn_selfconn_kernel(
    const float* __restrict__ node_feat,
    const float* __restrict__ sc_w0,
    const float* __restrict__ sc_w1,
    float* __restrict__ out, int Nn)
{
    __shared__ _Float16 s0T[32 * 32];   // sc_w0 transposed [v][u]
    __shared__ _Float16 s1T[32 * 32];   // sc_w1 transposed [v][u]

    int t = threadIdx.x;
    for (int i = t; i < 1024; i += 160) {
        int v = i >> 5, u = i & 31;
        s0T[v * 32 + u] = (_Float16)sc_w0[u * 32 + v];
        s1T[v * 32 + u] = (_Float16)sc_w1[u * 32 + v];
    }
    __syncthreads();

    int wave = t >> 5, lane = t & 31;
    int ln = lane & 15;
    int hi8 = (lane & 16) ? 8 : 0;
    int tile = blockIdx.x * 5 + wave;
    int n0r = tile * 16;
    if (n0r >= Nn) return;   // wave-uniform

    const float* xr = node_feat + (size_t)(n0r + ln) * 128;

    #pragma unroll
    for (int m = 0; m < 4; m++) {
        v16h a;
        if (m == 0) {
            a = gl_afrag_f32(xr, lane);               // A[n][u] = node_feat[n][u]
        } else {
            #pragma unroll
            for (int i = 0; i < 16; i++) {            // A[n][u] = node_feat[n][32+3u+(m-1)]
                int K = ((i < 8) ? i : i + 8) + ((lane & 16) ? 8 : 0);
                a[i] = (_Float16)xr[32 + 3 * K + (m - 1)];
            }
        }
        const _Float16* wT = (m == 0) ? s0T : s1T;
        #pragma unroll
        for (int nb = 0; nb < 2; nb++) {
            v16h b = lds_bfrag(wT, 32, nb, 0, lane);
            v8f c = {};
            c = wmma16(a, b, c);
            #pragma unroll
            for (int r = 0; r < 8; r++) {
                int n = n0r + r + hi8;
                int v = nb * 16 + ln;
                int f = (m == 0) ? v : (32 + 3 * v + (m - 1));
                out[(size_t)n * 128 + f] = c[r] * INV_SQRTM;
            }
        }
    }
}

// ---------------------------------------------------------------------------
// Kernel 2: edge MLP (3 WMMA GEMMs) + TFN message + atomic scatter-add.
// 256 threads = 8 waves; each wave owns one 16-edge tile (800000/16/8 = 6250 blocks).
// ---------------------------------------------------------------------------
__global__ __launch_bounds__(256) void tfn_edge_kernel(
    const int*   __restrict__ edge_index,   // [2][E]
    const float* __restrict__ node_feat,    // [N][128]
    const float* __restrict__ edge_feat,    // [E][4]
    const float* __restrict__ edge_embed,   // [E][32]
    const float* __restrict__ w1, const float* __restrict__ b1,   // [32][64],[64]
    const float* __restrict__ w2, const float* __restrict__ b2,   // [64][32],[32]
    const float* __restrict__ w3, const float* __restrict__ b3,   // [32][128],[128]
    float* __restrict__ out, int E)
{
    __shared__ _Float16 w1T[64 * 32];     // [n][k]
    __shared__ _Float16 w2T[32 * 64];
    __shared__ _Float16 w3T[128 * 32];
    __shared__ float sb1[64], sb2[32], sb3[128];
    __shared__ _Float16 hbuf[8][16 * 64]; // per-wave C->A transpose staging
    __shared__ int s_src[8][16], s_dst[8][16];
    __shared__ v4f s_ef[8][16];

    int t = threadIdx.x;
    for (int i = t; i < 64 * 32; i += 256) {
        int n = i >> 5, k = i & 31;
        w1T[n * 32 + k] = (_Float16)w1[k * 64 + n];
    }
    for (int i = t; i < 32 * 64; i += 256) {
        int n = i >> 6, k = i & 63;
        w2T[n * 64 + k] = (_Float16)w2[k * 32 + n];
    }
    for (int i = t; i < 128 * 32; i += 256) {
        int n = i >> 5, k = i & 31;
        w3T[n * 32 + k] = (_Float16)w3[k * 128 + n];
    }
    if (t < 64)  sb1[t] = b1[t];
    if (t < 32)  sb2[t] = b2[t];
    if (t < 128) sb3[t] = b3[t];
    __syncthreads();

    int wave = t >> 5, lane = t & 31;
    int ln = lane & 15;
    int hi8 = (lane & 16) ? 8 : 0;
    int tile = blockIdx.x * 8 + wave;
    int eb = tile * 16;
    if (eb >= E) return;   // wave-uniform -> EXEC stays all-ones for WMMA

    // Stage per-edge metadata (src, dst, edge_feat row) for this wave's tile.
    if (lane < 16) {
        int e = eb + lane;
        s_src[wave][lane] = edge_index[e];
        s_dst[wave][lane] = edge_index[E + e];
        s_ef[wave][lane]  = *(const v4f*)(edge_feat + (size_t)e * 4);
    }

    _Float16* hb = hbuf[wave];

    // ---- GEMM1: h1 = silu(edge_embed(16x32) @ W1(32x64) + b1) -> LDS f16 ----
    v16h a1 = gl_afrag_f32(edge_embed + (size_t)(eb + ln) * 32, lane);
    #pragma unroll
    for (int nb = 0; nb < 4; nb++) {
        v16h b = lds_bfrag(w1T, 32, nb, 0, lane);
        v8f c = {};
        c = wmma16(a1, b, c);
        float bias = sb1[nb * 16 + ln];
        #pragma unroll
        for (int r = 0; r < 8; r++) {
            float v = silu(c[r] + bias);
            hb[(r + hi8) * 64 + nb * 16 + ln] = (_Float16)v;  // h1[M][N]
        }
    }

    // ---- GEMM2: h2 = silu(h1(16x64) @ W2(64x32) + b2) -> LDS cols 0..31 ----
    v16h a2l = lds_afrag(hb, 64, 0,  lane);  // K 0..31
    v16h a2h = lds_afrag(hb, 64, 32, lane);  // K 32..63
    #pragma unroll
    for (int nb = 0; nb < 2; nb++) {
        v16h bl = lds_bfrag(w2T, 64, nb, 0,  lane);
        v16h bh = lds_bfrag(w2T, 64, nb, 32, lane);
        v8f c = {};
        c = wmma16(a2l, bl, c);
        c = wmma16(a2h, bh, c);
        float bias = sb2[nb * 16 + ln];
        #pragma unroll
        for (int r = 0; r < 8; r++) {
            float v = silu(c[r] + bias);
            hb[(r + hi8) * 64 + nb * 16 + ln] = (_Float16)v;  // h2[M][N], N<32
        }
    }

    // ---- GEMM3: w = h2(16x32) @ W3(32x128); keep 8 C-fragments in regs ----
    v16h a3 = lds_afrag(hb, 64, 0, lane);
    v8f c3[8];
    #pragma unroll
    for (int nb = 0; nb < 8; nb++) {
        v8f c = {};
        c3[nb] = wmma16(a3, lds_bfrag(w3T, 32, nb, 0, lane), c);
    }

    // ---- Message + scatter-add.  C-layout element (nb, r, lane) holds
    // (edge = r+hi8, u = nb*16 + lane%16); w000/w011/w101/w110 for the same
    // (edge,u) sit at the same position in fragments nbu, nbu+2, nbu+4, nbu+6.
    #pragma unroll
    for (int nbu = 0; nbu < 2; nbu++) {
        int u = nbu * 16 + ln;
        float bw000 = sb3[u], bw011 = sb3[32 + u];
        float bw101 = sb3[64 + u], bw110 = sb3[96 + u];
        #pragma unroll
        for (int r = 0; r < 8; r++) {
            int el = r + hi8;
            float w000 = c3[nbu][r]     + bw000;
            float w011 = c3[nbu + 2][r] + bw011;
            float w101 = c3[nbu + 4][r] + bw101;
            float w110 = c3[nbu + 6][r] + bw110;

            const float* xr = node_feat + (size_t)s_src[wave][el] * 128;
            v4f ef = s_ef[wave][el];                 // (e0, e1x, e1y, e1z)
            float x0 = xr[u];
            float xa = xr[32 + 3 * u];
            float xb = xr[33 + 3 * u];
            float xc = xr[34 + 3 * u];
            float dot = xa * ef.y + xb * ef.z + xc * ef.w;

            float o0  = S2F * w000 * x0 * ef.x + S2_3F * w110 * dot;
            float tq  = S2F * w011 * x0;
            float p1  = S2F * w101 * ef.x;
            float o1a = tq * ef.y + p1 * xa;
            float o1b = tq * ef.z + p1 * xb;
            float o1c = tq * ef.w + p1 * xc;

            float* orow = out + (size_t)s_dst[wave][el] * 128;
            atomicAdd(orow + u,             o0);
            atomicAdd(orow + 32 + 3 * u,    o1a);
            atomicAdd(orow + 33 + 3 * u,    o1b);
            atomicAdd(orow + 34 + 3 * u,    o1c);
        }
    }
}

// ---------------------------------------------------------------------------
extern "C" void kernel_launch(void* const* d_in, const int* in_sizes, int n_in,
                              void* d_out, int out_size, void* d_ws, size_t ws_size,
                              hipStream_t stream) {
    const int*   edge_index = (const int*)  d_in[0];
    const float* node_feat  = (const float*)d_in[1];
    const float* edge_feat  = (const float*)d_in[2];
    const float* edge_embed = (const float*)d_in[3];
    const float* fc_w1 = (const float*)d_in[4];
    const float* fc_b1 = (const float*)d_in[5];
    const float* fc_w2 = (const float*)d_in[6];
    const float* fc_b2 = (const float*)d_in[7];
    const float* fc_w3 = (const float*)d_in[8];
    const float* fc_b3 = (const float*)d_in[9];
    const float* sc_w0 = (const float*)d_in[10];
    const float* sc_w1 = (const float*)d_in[11];
    float* out = (float*)d_out;

    int E  = in_sizes[0] / 2;     // 800000
    int Nn = in_sizes[1] / 128;   // 50000

    // 1) self-connection: writes every out element (doubles as init)
    int nTiles = (Nn + 15) / 16;                 // 3125
    int nBlocks = (nTiles + 4) / 5;              // 625
    tfn_selfconn_kernel<<<nBlocks, 160, 0, stream>>>(node_feat, sc_w0, sc_w1, out, Nn);

    // 2) edge MLP + message + atomic scatter-add
    int eTiles = (E + 15) / 16;                  // 50000
    int eBlocks = (eTiles + 7) / 8;              // 6250
    tfn_edge_kernel<<<eBlocks, 256, 0, stream>>>(
        edge_index, node_feat, edge_feat, edge_embed,
        fc_w1, fc_b1, fc_w2, fc_b2, fc_w3, fc_b3, out, E);
}
